// MoE_32427003085359
// MI455X (gfx1250) — matrix operand order repaired and checked by reading, not compiled, not run on hardware
//
#include <hip/hip_runtime.h>
#include <stdint.h>

// ---- problem constants (match reference) ----
#define BTOK 4096
#define DDIM 1024
#define HDIM 2048
#define NEXP 8

// ---- tiling ----
#define BM 128
#define BN 128
#define BK 32
#define BSTRIDE (BK + 8)          // padded LDS stride in elements (80 B rows, 16B aligned)
#define BSTRIDE_U (BSTRIDE / 2)   // in uint (packed bf16 pairs)

typedef __attribute__((ext_vector_type(16))) __bf16 v16bf;
typedef __attribute__((ext_vector_type(8)))  float  v8f;

// ---------------- workspace layout (bytes) ----------------
// ctrl:     [0..7]=counts  [8..15]=cursors  [16..23]=offsets
static const size_t WS_CTRL    = 0;                          // 256 B
static const size_t WS_TIDX    = 256;                        // B*2 ints   = 32 KB
static const size_t WS_TGATE   = WS_TIDX  + BTOK*2*4;        // B*2 floats = 32 KB
static const size_t WS_MTOK    = WS_TGATE + BTOK*2*4;        // 2B ints
static const size_t WS_MSLOT   = WS_MTOK  + BTOK*2*4;        // 2B ints
static const size_t WS_MGATE   = WS_MSLOT + BTOK*2*4;        // 2B floats
static const size_t WS_XBF     = (WS_MGATE + BTOK*2*4 + 255) & ~(size_t)255; // B*D bf16 = 8 MB
static const size_t WS_HACT    = WS_XBF  + (size_t)BTOK*DDIM*2;              // 2B*H bf16 = 32 MB
static const size_t WS_YPART   = WS_HACT + (size_t)2*BTOK*HDIM*2;            // 2*B*D f32 = 32 MB

__device__ __forceinline__ unsigned pack_bf16(float lo, float hi) {
    union { __bf16 b[2]; unsigned u; } p;
    p.b[0] = (__bf16)lo;
    p.b[1] = (__bf16)hi;
    return p.u;
}

// ---------------- small kernels ----------------
__global__ void moe_zero_ctrl(int* ctrl) {
    if (threadIdx.x < 32) ctrl[threadIdx.x] = 0;
}

__global__ void moe_cvt_x(const float* __restrict__ x, __bf16* __restrict__ xbf) {
    const int n4 = BTOK * DDIM / 4;
    const float4* x4 = (const float4*)x;
    uint2* o = (uint2*)xbf;
    int i = blockIdx.x * blockDim.x + threadIdx.x;
    int stride = gridDim.x * blockDim.x;
    for (; i < n4; i += stride) {
        float4 v = x4[i];
        uint2 r;
        r.x = pack_bf16(v.x, v.y);
        r.y = pack_bf16(v.z, v.w);
        o[i] = r;
    }
}

// one wave (32 lanes) per token
__global__ __launch_bounds__(256) void moe_gate(const float* __restrict__ x,
                                                const float* __restrict__ wg,
                                                int* __restrict__ ctrl,
                                                int* __restrict__ tidx,
                                                float* __restrict__ tgate) {
    int wv   = threadIdx.x >> 5;
    int lane = threadIdx.x & 31;
    int tok  = blockIdx.x * 8 + wv;
    if (tok >= BTOK) return;

    float acc[NEXP];
#pragma unroll
    for (int e = 0; e < NEXP; e++) acc[e] = 0.f;
    const float* xr = x + (size_t)tok * DDIM;
    for (int d = lane; d < DDIM; d += 32) {
        float xv = xr[d];
        const float* wr = wg + (size_t)d * NEXP;
#pragma unroll
        for (int e = 0; e < NEXP; e++) acc[e] += xv * wr[e];
    }
#pragma unroll
    for (int e = 0; e < NEXP; e++)
#pragma unroll
        for (int off = 16; off > 0; off >>= 1) acc[e] += __shfl_xor(acc[e], off, 32);

    if (lane == 0) {
        int i0 = 0; float v0 = acc[0];
#pragma unroll
        for (int e = 1; e < NEXP; e++) if (acc[e] > v0) { v0 = acc[e]; i0 = e; }
        int i1 = (i0 == 0) ? 1 : 0; float v1 = acc[i1];
#pragma unroll
        for (int e = 0; e < NEXP; e++)
            if (e != i0 && acc[e] > v1) { v1 = acc[e]; i1 = e; }
        float e1 = __expf(v1 - v0);          // stable 2-way softmax, e0 = 1
        float inv = 1.f / (1.f + e1);
        tidx [2 * tok]     = i0;  tidx [2 * tok + 1] = i1;
        tgate[2 * tok]     = inv;
        tgate[2 * tok + 1] = e1 * inv;
        atomicAdd(&ctrl[i0], 1);
        atomicAdd(&ctrl[i1], 1);
    }
}

__global__ void moe_scan(int* ctrl) {
    if (threadIdx.x == 0 && blockIdx.x == 0) {
        int run = 0;
        for (int e = 0; e < NEXP; e++) {
            ctrl[16 + e] = run;   // offsets
            ctrl[8 + e]  = run;   // cursors
            run += ctrl[e];
        }
    }
}

__global__ void moe_assign(const int* __restrict__ tidx, const float* __restrict__ tgate,
                           int* __restrict__ ctrl,
                           int* __restrict__ mtok, int* __restrict__ mslot,
                           float* __restrict__ mgate) {
    int t = blockIdx.x * blockDim.x + threadIdx.x;
    if (t >= BTOK) return;
#pragma unroll
    for (int s = 0; s < 2; s++) {
        int e = tidx[2 * t + s];
        int pos = atomicAdd(&ctrl[8 + e], 1);
        mtok[pos]  = t;
        mslot[pos] = s;
        mgate[pos] = tgate[2 * t + s];
    }
}

// ---------------- WMMA helpers ----------------
union FragBF { v16bf v; uint4 q[2]; };

__device__ __forceinline__ void async_b128(unsigned ldsaddr, unsigned long long gaddr) {
    asm volatile("global_load_async_to_lds_b128 %0, %1, off" :: "v"(ldsaddr), "v"(gaddr) : "memory");
}
__device__ __forceinline__ void async_b128_o16(unsigned ldsaddr, unsigned long long gaddr) {
    asm volatile("global_load_async_to_lds_b128 %0, %1, off offset:16" :: "v"(ldsaddr), "v"(gaddr) : "memory");
}
__device__ __forceinline__ void wait_async0() {
    asm volatile("s_wait_asynccnt 0" ::: "memory");
}

// stage a BKxBN f32 weight tile transposed+converted into LDS as [n][k] packed bf16 pairs.
// lane <-> n adjacent: coalesced b32 global loads, conflict-free b32 DS stores (bank stride 20).
__device__ __forceinline__ void stage_w_tile(const float* __restrict__ Wk, int ldw,
                                             unsigned* __restrict__ shBu, int tid) {
    int nn = tid & 127;
    int kg = (tid >> 7) * 16;               // k-group 0 or 16
    const float* wp = Wk + (size_t)kg * ldw + nn;
    unsigned* dst = shBu + nn * BSTRIDE_U + (kg >> 1);
#pragma unroll
    for (int i = 0; i < 8; i++) {
        float lo = wp[0];
        float hi = wp[ldw];
        dst[i] = pack_bf16(lo, hi);
        wp += 2 * (size_t)ldw;
    }
}

// ---------------- fused GEMM body ----------------
// Wave grid 4(m) x 2(n): each wave computes 32 rows x 64 cols -> acc[2][4] (8 WMMA tiles).
#define GEMM_COMPUTE_STEP(shA_, shBu_, mw, nw, lh, lm, acc)                                   \
    {                                                                                         \
        FragBF af0, af1;                                                                      \
        const __bf16* arow0 = &shA_[((mw) * 32 + (lm)) * BK + (lh) * 8];                      \
        const __bf16* arow1 = arow0 + 16 * BK;                                                \
        af0.q[0] = *(const uint4*)(arow0);                                                    \
        af0.q[1] = *(const uint4*)(arow0 + 16);                                               \
        af1.q[0] = *(const uint4*)(arow1);                                                    \
        af1.q[1] = *(const uint4*)(arow1 + 16);                                               \
        _Pragma("unroll")                                                                     \
        for (int b = 0; b < 4; b++) {                                                         \
            FragBF bfr;                                                                       \
            const char* bp = (const char*)shBu_ + ((nw) * 64 + b * 16 + (lm)) * (BSTRIDE * 2) \
                           + (lh) * 16;                                                       \
            bfr.q[0] = *(const uint4*)(bp);                                                   \
            bfr.q[1] = *(const uint4*)(bp + 32);                                              \
            acc[0][b] = __builtin_amdgcn_wmma_f32_16x16x32_bf16(                              \
                false, af0.v, false, bfr.v, (short)0, acc[0][b], false, false);               \
            acc[1][b] = __builtin_amdgcn_wmma_f32_16x16x32_bf16(                              \
                false, af1.v, false, bfr.v, (short)0, acc[1][b], false, false);               \
        }                                                                                     \
    }

// ---------------- fc1: Hact = relu(X[tok] @ W1[e] + b1[e]) in bf16 ----------------
__global__ __launch_bounds__(256) void moe_fc1(const __bf16* __restrict__ xbf,
                                               const float* __restrict__ W1,
                                               const float* __restrict__ b1,
                                               const int* __restrict__ ctrl,
                                               const int* __restrict__ mtok,
                                               __bf16* __restrict__ Hact) {
    int e = blockIdx.z, mt = blockIdx.y, nt = blockIdx.x;
    int cnt  = ctrl[e];
    int base = ctrl[16 + e];
    int mbase = mt * BM;
    if (mbase >= cnt) return;
    int rows = cnt - mbase; if (rows > BM) rows = BM;
    int n0 = nt * BN;
    const float* W1e = W1 + (size_t)e * DDIM * HDIM;
    const float* b1e = b1 + (size_t)e * HDIM;

    __shared__ int sh_tok[BM];
    __shared__ __align__(16) __bf16   shA[BM * BK];
    __shared__ __align__(16) unsigned shBu[BN * BSTRIDE_U];

    int tid = threadIdx.x;
    if (tid < BM) {
        int r = tid < rows ? tid : rows - 1;
        sh_tok[tid] = mtok[base + mbase + r];
    }
    __syncthreads();

    int rA = tid >> 1;
    int halfB = (tid & 1) * 32;   // byte offset in 64B row segment
    unsigned ldsA = (unsigned)(uintptr_t)(&shA[0]) + (unsigned)(rA * 64 + halfB);
    unsigned long long gA = (unsigned long long)(uintptr_t)xbf
                          + (unsigned long long)sh_tok[rA] * (DDIM * 2) + (unsigned)halfB;

    int wv = tid >> 5, lane = tid & 31;
    int mw = wv & 3, nw = wv >> 2;
    int lh = lane >> 4, lm = lane & 15;
    v8f acc[2][4] = {};

    for (int k0 = 0; k0 < DDIM; k0 += BK) {
        async_b128(ldsA, gA);          // gathered bf16 token rows -> LDS
        async_b128_o16(ldsA, gA);
        stage_w_tile(W1e + (size_t)k0 * HDIM + n0, HDIM, shBu, tid);
        wait_async0();
        __syncthreads();
        GEMM_COMPUTE_STEP(shA, shBu, mw, nw, lh, lm, acc);
        __syncthreads();
        gA += BK * 2;
    }

#pragma unroll
    for (int b = 0; b < 4; b++) {
        int n = n0 + nw * 64 + b * 16 + lm;
        float bb = b1e[n];
#pragma unroll
        for (int a = 0; a < 2; a++)
#pragma unroll
            for (int i = 0; i < 8; i++) {
                int gr = mw * 32 + a * 16 + lh * 8 + i;
                if (gr < rows) {
                    float h = acc[a][b][i] + bb;
                    h = h > 0.f ? h : 0.f;
                    Hact[(size_t)(base + mbase + gr) * HDIM + n] = (__bf16)h;
                }
            }
    }
}

// ---------------- fc2: y_part[slot,tok] = gate * (Hact @ W2[e] + b2[e]) ----------------
__global__ __launch_bounds__(256) void moe_fc2(const __bf16* __restrict__ Hact,
                                               const float* __restrict__ W2,
                                               const float* __restrict__ b2,
                                               const int* __restrict__ ctrl,
                                               const int* __restrict__ mtok,
                                               const int* __restrict__ mslot,
                                               const float* __restrict__ mgate,
                                               float* __restrict__ ypart) {
    int e = blockIdx.z, mt = blockIdx.y, nt = blockIdx.x;
    int cnt  = ctrl[e];
    int base = ctrl[16 + e];
    int mbase = mt * BM;
    if (mbase >= cnt) return;
    int rows = cnt - mbase; if (rows > BM) rows = BM;
    int n0 = nt * BN;
    const float* W2e = W2 + (size_t)e * HDIM * DDIM;
    const float* b2e = b2 + (size_t)e * DDIM;

    __shared__ int   sh_tok[BM];
    __shared__ int   sh_slot[BM];
    __shared__ float sh_gate[BM];
    __shared__ __align__(16) __bf16   shA[BM * BK];
    __shared__ __align__(16) unsigned shBu[BN * BSTRIDE_U];

    int tid = threadIdx.x;
    if (tid < BM) {
        int r = tid < rows ? tid : rows - 1;
        int pos = base + mbase + r;
        sh_tok[tid]  = mtok[pos];
        sh_slot[tid] = mslot[pos];
        sh_gate[tid] = mgate[pos];
    }
    __syncthreads();

    int rA = tid >> 1;
    int halfB = (tid & 1) * 32;
    int rowA = rA < rows ? rA : rows - 1;
    unsigned ldsA = (unsigned)(uintptr_t)(&shA[0]) + (unsigned)(rA * 64 + halfB);
    unsigned long long gA = (unsigned long long)(uintptr_t)Hact
                          + (unsigned long long)(base + mbase + rowA) * (HDIM * 2) + (unsigned)halfB;

    int wv = tid >> 5, lane = tid & 31;
    int mw = wv & 3, nw = wv >> 2;
    int lh = lane >> 4, lm = lane & 15;
    v8f acc[2][4] = {};

    for (int k0 = 0; k0 < HDIM; k0 += BK) {
        async_b128(ldsA, gA);
        async_b128_o16(ldsA, gA);
        stage_w_tile(W2e + (size_t)k0 * DDIM + n0, DDIM, shBu, tid);
        wait_async0();
        __syncthreads();
        GEMM_COMPUTE_STEP(shA, shBu, mw, nw, lh, lm, acc);
        __syncthreads();
        gA += BK * 2;
    }

#pragma unroll
    for (int b = 0; b < 4; b++) {
        int n = n0 + nw * 64 + b * 16 + lm;
        float bb = b2e[n];
#pragma unroll
        for (int a = 0; a < 2; a++)
#pragma unroll
            for (int i = 0; i < 8; i++) {
                int gr = mw * 32 + a * 16 + lh * 8 + i;
                if (gr < rows) {
                    int   tok  = sh_tok[gr];
                    int   slot = sh_slot[gr];
                    float g    = sh_gate[gr];
                    float val  = g * (acc[a][b][i] + bb);
                    ypart[((size_t)slot * BTOK + tok) * DDIM + n] = val;
                }
            }
    }
}

__global__ void moe_combine(const float* __restrict__ ypart, float* __restrict__ out) {
    const int N4 = (BTOK * DDIM) / 4;
    const float4* y0 = (const float4*)ypart;
    const float4* y1 = y0 + N4;
    float4* o = (float4*)out;
    int i = blockIdx.x * blockDim.x + threadIdx.x;
    int stride = gridDim.x * blockDim.x;
    for (; i < N4; i += stride) {
        float4 a = y0[i], b = y1[i], r;
        r.x = a.x + b.x; r.y = a.y + b.y; r.z = a.z + b.z; r.w = a.w + b.w;
        o[i] = r;
    }
}

// ---------------- launch ----------------
extern "C" void kernel_launch(void* const* d_in, const int* in_sizes, int n_in,
                              void* d_out, int out_size, void* d_ws, size_t ws_size,
                              hipStream_t stream) {
    (void)in_sizes; (void)n_in; (void)out_size; (void)ws_size;
    const float* x  = (const float*)d_in[0];
    const float* wg = (const float*)d_in[1];
    const float* W1 = (const float*)d_in[2];
    const float* b1 = (const float*)d_in[3];
    const float* W2 = (const float*)d_in[4];
    const float* b2 = (const float*)d_in[5];
    float* out = (float*)d_out;

    char* ws = (char*)d_ws;
    int*    ctrl  = (int*)  (ws + WS_CTRL);
    int*    tidx  = (int*)  (ws + WS_TIDX);
    float*  tgate = (float*)(ws + WS_TGATE);
    int*    mtok  = (int*)  (ws + WS_MTOK);
    int*    mslot = (int*)  (ws + WS_MSLOT);
    float*  mgate = (float*)(ws + WS_MGATE);
    __bf16* xbf   = (__bf16*)(ws + WS_XBF);
    __bf16* Hact  = (__bf16*)(ws + WS_HACT);
    float*  ypart = (float*) (ws + WS_YPART);

    moe_zero_ctrl<<<1, 64, 0, stream>>>(ctrl);
    moe_cvt_x<<<1024, 256, 0, stream>>>(x, xbf);
    moe_gate<<<BTOK / 8, 256, 0, stream>>>(x, wg, ctrl, tidx, tgate);
    moe_scan<<<1, 1, 0, stream>>>(ctrl);
    moe_assign<<<BTOK / 256, 256, 0, stream>>>(tidx, tgate, ctrl, mtok, mslot, mgate);
    moe_fc1<<<dim3(HDIM / BN, BTOK / BM, NEXP), 256, 0, stream>>>(xbf, W1, b1, ctrl, mtok, Hact);
    moe_fc2<<<dim3(DDIM / BN, BTOK / BM, NEXP), 256, 0, stream>>>(Hact, W2, b2, ctrl, mtok, mslot, mgate, ypart);
    moe_combine<<<1024, 256, 0, stream>>>(ypart, out);
}
